// RandMultiHeadAttention_13975823581898
// MI455X (gfx1250) — matrix operand order repaired and checked by reading, not compiled, not run on hardware
//
#include <hip/hip_runtime.h>

typedef __attribute__((ext_vector_type(16))) __bf16 v16bf;
typedef __attribute__((ext_vector_type(8)))  float  v8f;

constexpr int Bc = 4, Sc = 2048, Ec = 512, Hc = 8, Mc = 256, Dc = 64;
constexpr int BH    = Bc * Hc;      // 32
constexpr int NTOK  = Bc * Sc;      // 8192
constexpr int CHUNK = 128;
constexpr int NC    = Sc / CHUNK;   // 16
constexpr int LDAM  = CHUNK + 8;    // padded LDS leading dim

// ---------------- WMMA helpers (CDNA5 16x16x32 bf16, fp32 accum) ----------------

__device__ inline v8f wmma_bf16(v16bf a, v16bf b, v8f c) {
  return __builtin_amdgcn_wmma_f32_16x16x32_bf16(false, a, false, b, (short)0, c,
                                                 false, false);
}

// A fragment: A row-major 16x32 tile at base, leading dim ld (in elements).
// Contiguous k order used here: lanes 0-15 rows M=0..15 K=0..15; lanes 16-31 K=16..31
__device__ inline v16bf frag_a_row(const __bf16* A, int ld) {
  int lane = threadIdx.x & 31;
  int m  = lane & 15;
  int kh = (lane >> 4) << 4;                 // 0 or 16
  const __bf16* p = A + (size_t)m * ld + kh;
  v16bf r;
#pragma unroll
  for (int e = 0; e < 8; ++e) r[e] = p[e];
#pragma unroll
  for (int e = 0; e < 8; ++e) r[8 + e] = p[8 + e];
  return r;
}

// B fragment from transposed storage Bt[n][k] (rows of B^T contiguous in k).
// Contiguous k order used here: lanes 0-15 col n, K=0..15; lanes 16-31 col n, K=16..31
__device__ inline v16bf frag_b_col(const __bf16* Bt, int ld) {
  int lane = threadIdx.x & 31;
  int n  = lane & 15;
  int k0 = (lane >> 4) << 4;                 // 0 or 16
  const __bf16* p = Bt + (size_t)n * ld + k0;
  v16bf r;
#pragma unroll
  for (int e = 0; e < 16; ++e) r[e] = p[e];
  return r;
}

// B fragment from row-major B[k][n] (strided gather over k).
__device__ inline v16bf frag_b_row(const __bf16* Bm, int ld) {
  int lane = threadIdx.x & 31;
  int n  = lane & 15;
  int k0 = (lane >> 4) << 4;
  v16bf r;
#pragma unroll
  for (int e = 0; e < 16; ++e) r[e] = Bm[(size_t)(k0 + e) * ld + n];
  return r;
}

// ---------------- elementwise / prep kernels ----------------

__global__ void k_f32_to_bf16(const float* __restrict__ x, __bf16* __restrict__ y, int n) {
  int i = blockIdx.x * blockDim.x + threadIdx.x;
  if (i < n) y[i] = (__bf16)x[i];
}

// Y[rows x N] = Xbf[rows x K] @ Wbf^T (+ bias).  Wbf row-major [N x K].
// 2x2 register-blocked: one 32x32 macro tile per wave.
// rows,N mult of 32; K mult of 32.
__global__ void k_gemm_xwt(const __bf16* __restrict__ X, const __bf16* __restrict__ W,
                           const float* __restrict__ bias, float* __restrict__ Y,
                           int rows, int N, int K) {
  int wave = threadIdx.x >> 5;
  int mt = blockIdx.x * (blockDim.x >> 5) + wave;
  int ntc = N >> 5;
  int tm = mt / ntc;
  int tn = mt - tm * ntc;
  if (tm >= (rows >> 5)) return;               // wave-uniform
  const __bf16* A0 = X + (size_t)tm * 32 * K;
  const __bf16* A1 = A0 + (size_t)16 * K;
  const __bf16* B0 = W + (size_t)tn * 32 * K;
  const __bf16* B1 = B0 + (size_t)16 * K;
  v8f acc00 = {}, acc01 = {}, acc10 = {}, acc11 = {};
  for (int k0 = 0; k0 < K; k0 += 32) {
    if (k0 + 64 < K) {                          // prefetch streamed A operand
      __builtin_prefetch(A0 + k0 + 64, 0, 0);
      __builtin_prefetch(A1 + k0 + 64, 0, 0);
    }
    v16bf a0 = frag_a_row(A0 + k0, K);
    v16bf a1 = frag_a_row(A1 + k0, K);
    v16bf b0 = frag_b_col(B0 + k0, K);
    v16bf b1 = frag_b_col(B1 + k0, K);
    acc00 = wmma_bf16(a0, b0, acc00);
    acc01 = wmma_bf16(a0, b1, acc01);
    acc10 = wmma_bf16(a1, b0, acc10);
    acc11 = wmma_bf16(a1, b1, acc11);
  }
  int lane = threadIdx.x & 31;
  int n = lane & 15, mb = (lane >> 4) << 3;
  float bv0 = bias ? bias[tn * 32 + n] : 0.0f;
  float bv1 = bias ? bias[tn * 32 + 16 + n] : 0.0f;
  float* Y0 = Y + (size_t)(tm * 32) * N + tn * 32;
  float* Y1 = Y0 + (size_t)16 * N;
#pragma unroll
  for (int r = 0; r < 8; ++r) {
    Y0[(size_t)(mb + r) * N + n]      = acc00[r] + bv0;
    Y0[(size_t)(mb + r) * N + 16 + n] = acc01[r] + bv1;
    Y1[(size_t)(mb + r) * N + n]      = acc10[r] + bv0;
    Y1[(size_t)(mb + r) * N + 16 + n] = acc11[r] + bv1;
  }
}

// q/k [B,S,E] fp32 -> normalized bf16 head-major [BH,S,D] + 0.5*||x||^2 per row
__global__ void k_prep_feat(const float* __restrict__ X, __bf16* __restrict__ Xn,
                            float* __restrict__ sq, float normalizer) {
  int i = blockIdx.x * blockDim.x + threadIdx.x;   // (b,h,s) row
  if (i >= BH * Sc) return;
  int s = i % Sc;
  int bh = i / Sc;
  int h = bh % Hc, b = bh / Hc;
  const float* src = X + ((size_t)b * Sc + s) * Ec + h * Dc;
  __bf16* dst = Xn + ((size_t)bh * Sc + s) * Dc;
  float acc = 0.0f;
#pragma unroll 8
  for (int d = 0; d < Dc; ++d) {
    float v = src[d] * normalizer;
    dst[d] = (__bf16)v;
    acc += v * v;
  }
  sq[i] = 0.5f * acc;
}

// v [B,S,E] fp32 -> bf16 head-major [BH,S,D]
__global__ void k_split_v(const float* __restrict__ vf, __bf16* __restrict__ vb) {
  int i = blockIdx.x * blockDim.x + threadIdx.x;
  if (i >= Bc * Sc * Ec) return;
  int d = i % Dc;
  int s = (i / Dc) % Sc;
  int h = (i / (Dc * Sc)) % Hc;
  int b = i / (Dc * Sc * Hc);
  vb[i] = (__bf16)vf[((size_t)b * Sc + s) * Ec + h * Dc + d];
}

// FAVOR+ features: one wave per row of proj [BHS, M]; row-max stabilized exp.
// phi  : [BH,S,M] bf16 ;  phiT (optional) : [BH,M,S] bf16
__global__ void k_phi(const float* __restrict__ proj, const float* __restrict__ sq,
                      __bf16* __restrict__ phi, __bf16* __restrict__ phiT) {
  int row = blockIdx.x * (blockDim.x >> 5) + (threadIdx.x >> 5);
  if (row >= BH * Sc) return;
  int lane = threadIdx.x & 31;
  const float* p = proj + (size_t)row * Mc;
  float sqv = sq[row];
  float vals[8];
  float mx = -3.0e38f;
#pragma unroll
  for (int j = 0; j < 8; ++j) {
    vals[j] = p[lane + j * 32] - sqv;
    mx = fmaxf(mx, vals[j]);
  }
#pragma unroll
  for (int off = 16; off >= 1; off >>= 1) mx = fmaxf(mx, __shfl_xor(mx, off, 32));
  int s = row % Sc;
  int bh = row / Sc;
  __bf16* out = phi + (size_t)row * Mc;
#pragma unroll
  for (int j = 0; j < 8; ++j) {
    int m = lane + j * 32;
    float f = __expf(vals[j] - mx) * 0.0625f + 1e-6f;   // /sqrt(256)
    out[m] = (__bf16)f;
    if (phiT) phiT[((size_t)bh * Mc + m) * Sc + s] = (__bf16)f;
  }
}

// chunk-local state Skv_c = phiK_c^T @ V_c  ([M,D] fp32) and z_c[m] = sum_t phiK.
// grid: BH*NC blocks, 256 threads. 8 waves; each wave: 2 tm x 4 tn tiles.
__global__ void k_chunk_state(const __bf16* __restrict__ phiKT, const __bf16* __restrict__ vb,
                              float* __restrict__ states, float* __restrict__ zst) {
  int blk = blockIdx.x;
  int c = blk % NC, bh = blk / NC;
  int s0 = c * CHUNK;
  const __bf16* At = phiKT + (size_t)bh * Mc * Sc + s0;   // A[m][t] contiguous in t
  const __bf16* Bv = vb + ((size_t)bh * Sc + s0) * Dc;    // B[t][d] row-major
  float* out = states + ((size_t)bh * NC + c) * (Mc * Dc);
  int wave = threadIdx.x >> 5;
  int lane = threadIdx.x & 31;
  int n = lane & 15, mb = (lane >> 4) << 3;
  int tm0 = wave * 2, tm1 = wave * 2 + 1;
  v8f acc[8];
#pragma unroll
  for (int j = 0; j < 8; ++j) acc[j] = (v8f){};
  for (int k0 = 0; k0 < CHUNK; k0 += 32) {
    v16bf a0 = frag_a_row(At + (size_t)tm0 * 16 * Sc + k0, Sc);
    v16bf a1 = frag_a_row(At + (size_t)tm1 * 16 * Sc + k0, Sc);
#pragma unroll
    for (int j = 0; j < 4; ++j) {
      v16bf b = frag_b_row(Bv + (size_t)k0 * Dc + j * 16, Dc);
      acc[j]     = wmma_bf16(a0, b, acc[j]);
      acc[4 + j] = wmma_bf16(a1, b, acc[4 + j]);
    }
  }
#pragma unroll
  for (int j = 0; j < 4; ++j) {
    float* C0 = out + (size_t)(tm0 * 16) * Dc + j * 16;
    float* C1 = out + (size_t)(tm1 * 16) * Dc + j * 16;
#pragma unroll
    for (int r = 0; r < 8; ++r) {
      C0[(size_t)(mb + r) * Dc + n] = acc[j][r];
      C1[(size_t)(mb + r) * Dc + n] = acc[4 + j][r];
    }
  }
  int m = threadIdx.x;                     // 256 threads == M
  const __bf16* pk = phiKT + ((size_t)bh * Mc + m) * Sc + s0;
  float z = 0.0f;
  for (int t = 0; t < CHUNK; ++t) z += (float)pk[t];
  zst[((size_t)bh * NC + c) * Mc + m] = z;
}

// exclusive prefix over chunks for Skv (bf16 out) and z (fp32 out)
__global__ void k_prefix(const float* __restrict__ states, const float* __restrict__ zst,
                         __bf16* __restrict__ pst, float* __restrict__ pz) {
  int i = blockIdx.x * blockDim.x + threadIdx.x;
  if (i >= BH * Mc * Dc) return;
  int bh = i / (Mc * Dc);
  int e = i % (Mc * Dc);
  float run = 0.0f;
  for (int c = 0; c < NC; ++c) {
    size_t idx = ((size_t)bh * NC + c) * (Mc * Dc) + e;
    pst[idx] = (__bf16)run;
    run += states[idx];
  }
  if (e < Mc) {
    float rz = 0.0f;
    for (int c = 0; c < NC; ++c) {
      size_t idx = ((size_t)bh * NC + c) * Mc + e;
      pz[idx] = rz;
      rz += zst[idx];
    }
  }
}

// per-chunk output:
//  A = mask(phiQ_c @ phiK_c^T)      (WMMA, K=256, -> LDS bf16)
//  O = phiQ_c @ Skv_prefix + A @ V_c (WMMA, K=256 + K=128)
//  den[t] = rowsum(A[t]) + phiQ[t].z_prefix ; attn = O/den  -> [B,S,E] fp32
__global__ void k_chunk_out(const __bf16* __restrict__ phiQ, const __bf16* __restrict__ phiK,
                            const __bf16* __restrict__ vb, const __bf16* __restrict__ pst,
                            const float* __restrict__ pz, float* __restrict__ attn) {
  __shared__ __bf16 Am[CHUNK][LDAM];
  __shared__ float den_sh[CHUNK];
  int blk = blockIdx.x;
  int c = blk % NC, bh = blk / NC;
  int h = bh % Hc, b = bh / Hc;
  int s0 = c * CHUNK;
  const __bf16* Qc = phiQ + ((size_t)bh * Sc + s0) * Mc;  // [CHUNK][M]
  const __bf16* Kc = phiK + ((size_t)bh * Sc + s0) * Mc;  // [CHUNK][M]
  const __bf16* Vc = vb + ((size_t)bh * Sc + s0) * Dc;    // [CHUNK][D]
  const __bf16* Pc = pst + ((size_t)bh * NC + c) * (Mc * Dc); // [M][D]
  const float*  Zc = pz + ((size_t)bh * NC + c) * Mc;
  int wave = threadIdx.x >> 5;
  int lane = threadIdx.x & 31;
  int n = lane & 15, mb = (lane >> 4) << 3;

  // 1) masked A -> LDS. Wave handles row-of-tiles tm = wave, tn = 0..7;
  //    A fragment hoisted and reused across the 8 column tiles.
  {
    v8f acc[8];
#pragma unroll
    for (int j = 0; j < 8; ++j) acc[j] = (v8f){};
    for (int k0 = 0; k0 < Mc; k0 += 32) {
      v16bf a = frag_a_row(Qc + (size_t)wave * 16 * Mc + k0, Mc);
#pragma unroll
      for (int j = 0; j < 8; ++j) {
        v16bf bb = frag_b_col(Kc + (size_t)j * 16 * Mc + k0, Mc);
        acc[j] = wmma_bf16(a, bb, acc[j]);
      }
    }
#pragma unroll
    for (int j = 0; j < 8; ++j) {
#pragma unroll
      for (int r = 0; r < 8; ++r) {
        int t = wave * 16 + mb + r;
        int t2 = j * 16 + n;
        Am[t][t2] = (t2 <= t) ? (__bf16)acc[j][r] : (__bf16)0.0f;
      }
    }
  }
  __syncthreads();

  // 2) denominators
  if (threadIdx.x < CHUNK) {
    int t = threadIdx.x;
    float rs = 0.0f;
    for (int t2 = 0; t2 <= t; ++t2) rs += (float)Am[t][t2];
    const __bf16* qrow = Qc + (size_t)t * Mc;
    float dot = 0.0f;
#pragma unroll 8
    for (int m = 0; m < Mc; ++m) dot += (float)qrow[m] * Zc[m];
    den_sh[t] = rs + dot + 1e-6f;
  }
  __syncthreads();

  // 3) O = inter + intra. Wave handles tm = wave, tn = 0..3; A hoisted.
  {
    v8f acc[4];
#pragma unroll
    for (int j = 0; j < 4; ++j) acc[j] = (v8f){};
    for (int k0 = 0; k0 < Mc; k0 += 32) {         // inter: phiQ @ Skv_prefix
      v16bf a = frag_a_row(Qc + (size_t)wave * 16 * Mc + k0, Mc);
#pragma unroll
      for (int j = 0; j < 4; ++j) {
        v16bf bb = frag_b_row(Pc + (size_t)k0 * Dc + j * 16, Dc);
        acc[j] = wmma_bf16(a, bb, acc[j]);
      }
    }
    for (int k0 = 0; k0 < CHUNK; k0 += 32) {      // intra: A_masked @ V
      v16bf a = frag_a_row(&Am[wave * 16][0] + k0, LDAM);
#pragma unroll
      for (int j = 0; j < 4; ++j) {
        v16bf bb = frag_b_row(Vc + (size_t)k0 * Dc + j * 16, Dc);
        acc[j] = wmma_bf16(a, bb, acc[j]);
      }
    }
#pragma unroll
    for (int j = 0; j < 4; ++j) {
#pragma unroll
      for (int r = 0; r < 8; ++r) {
        int t = wave * 16 + mb + r;
        int d = j * 16 + n;
        attn[((size_t)b * Sc + s0 + t) * Ec + h * Dc + d] = acc[j][r] / den_sh[t];
      }
    }
  }
}

// ---------------- host launcher ----------------

static inline size_t al256(size_t x) { return (x + 255) & ~(size_t)255; }

extern "C" void kernel_launch(void* const* d_in, const int* in_sizes, int n_in,
                              void* d_out, int out_size, void* d_ws, size_t ws_size,
                              hipStream_t stream) {
  (void)in_sizes; (void)n_in; (void)out_size; (void)ws_size;
  const float* query = (const float*)d_in[0];
  const float* key_  = (const float*)d_in[1];
  const float* value = (const float*)d_in[2];
  const float* Wq = (const float*)d_in[3];
  const float* Wk = (const float*)d_in[4];
  const float* Wv = (const float*)d_in[5];
  const float* W0 = (const float*)d_in[6];
  const float* bq = (const float*)d_in[7];
  const float* bk = (const float*)d_in[8];
  const float* bv = (const float*)d_in[9];
  const float* b0 = (const float*)d_in[10];
  const float* pm = (const float*)d_in[11];
  float* out = (float*)d_out;

  char* p = (char*)d_ws;
  auto carve = [&](size_t bytes) -> void* { void* r = (void*)p; p += al256(bytes); return r; };

  __bf16* Xbf  = (__bf16*)carve((size_t)NTOK * Ec * 2);
  __bf16* Wqb  = (__bf16*)carve((size_t)Ec * Ec * 2);
  __bf16* Wkb  = (__bf16*)carve((size_t)Ec * Ec * 2);
  __bf16* Wvb  = (__bf16*)carve((size_t)Ec * Ec * 2);
  __bf16* W0b  = (__bf16*)carve((size_t)Ec * Ec * 2);
  __bf16* pmb  = (__bf16*)carve((size_t)Mc * Dc * 2);
  float*  qf   = (float*)carve((size_t)NTOK * Ec * 4);
  float*  kf   = (float*)carve((size_t)NTOK * Ec * 4);
  float*  vf   = (float*)carve((size_t)NTOK * Ec * 4);
  __bf16* qn   = (__bf16*)carve((size_t)BH * Sc * Dc * 2);
  __bf16* kn   = (__bf16*)carve((size_t)BH * Sc * Dc * 2);
  __bf16* vb   = (__bf16*)carve((size_t)BH * Sc * Dc * 2);
  float*  sqq  = (float*)carve((size_t)BH * Sc * 4);
  float*  sqk  = (float*)carve((size_t)BH * Sc * 4);
  float*  prj  = (float*)carve((size_t)BH * Sc * Mc * 4);
  __bf16* phq  = (__bf16*)carve((size_t)BH * Sc * Mc * 2);
  __bf16* phk  = (__bf16*)carve((size_t)BH * Sc * Mc * 2);
  __bf16* phkT = (__bf16*)carve((size_t)BH * Mc * Sc * 2);
  float*  stf  = (float*)carve((size_t)BH * NC * Mc * Dc * 4);
  float*  zst  = (float*)carve((size_t)BH * NC * Mc * 4);
  __bf16* pst  = (__bf16*)carve((size_t)BH * NC * Mc * Dc * 2);
  float*  pz   = (float*)carve((size_t)BH * NC * Mc * 4);
  float*  attn = (float*)carve((size_t)NTOK * Ec * 4);

  const int T = 256;
  const float normalizer = 0.35355339059327373f;   // 1/sqrt(sqrt(64))
  int gemm_blocks = (NTOK / 32) * (Ec / 32) / 8;    // 512  (32x32 macro tiles)
  int proj_blocks = (BH * Sc / 32) * (Mc / 32) / 8; // 2048

  // weight / pm conversion
  k_f32_to_bf16<<<(Ec * Ec + T - 1) / T, T, 0, stream>>>(Wq, Wqb, Ec * Ec);
  k_f32_to_bf16<<<(Ec * Ec + T - 1) / T, T, 0, stream>>>(Wk, Wkb, Ec * Ec);
  k_f32_to_bf16<<<(Ec * Ec + T - 1) / T, T, 0, stream>>>(Wv, Wvb, Ec * Ec);
  k_f32_to_bf16<<<(Ec * Ec + T - 1) / T, T, 0, stream>>>(W0, W0b, Ec * Ec);
  k_f32_to_bf16<<<(Mc * Dc + T - 1) / T, T, 0, stream>>>(pm, pmb, Mc * Dc);

  // input projections
  k_f32_to_bf16<<<NTOK * Ec / T, T, 0, stream>>>(query, Xbf, NTOK * Ec);
  k_gemm_xwt<<<gemm_blocks, T, 0, stream>>>(Xbf, Wqb, bq, qf, NTOK, Ec, Ec);
  k_f32_to_bf16<<<NTOK * Ec / T, T, 0, stream>>>(key_, Xbf, NTOK * Ec);
  k_gemm_xwt<<<gemm_blocks, T, 0, stream>>>(Xbf, Wkb, bk, kf, NTOK, Ec, Ec);
  k_f32_to_bf16<<<NTOK * Ec / T, T, 0, stream>>>(value, Xbf, NTOK * Ec);
  k_gemm_xwt<<<gemm_blocks, T, 0, stream>>>(Xbf, Wvb, bv, vf, NTOK, Ec, Ec);

  // feature prep
  k_prep_feat<<<BH * Sc / T, T, 0, stream>>>(qf, qn, sqq, normalizer);
  k_prep_feat<<<BH * Sc / T, T, 0, stream>>>(kf, kn, sqk, normalizer);
  k_split_v<<<Bc * Sc * Ec / T, T, 0, stream>>>(vf, vb);

  // phi_q / phi_k (proj GEMM + exp features); prj buffer reused
  k_gemm_xwt<<<proj_blocks, T, 0, stream>>>(qn, pmb, nullptr, prj, BH * Sc, Mc, Dc);
  k_phi<<<BH * Sc / 8, T, 0, stream>>>(prj, sqq, phq, nullptr);
  k_gemm_xwt<<<proj_blocks, T, 0, stream>>>(kn, pmb, nullptr, prj, BH * Sc, Mc, Dc);
  k_phi<<<BH * Sc / 8, T, 0, stream>>>(prj, sqk, phk, phkT);

  // chunked causal scan
  k_chunk_state<<<BH * NC, T, 0, stream>>>(phkT, vb, stf, zst);
  k_prefix<<<BH * Mc * Dc / T, T, 0, stream>>>(stf, zst, pst, pz);
  k_chunk_out<<<BH * NC, T, 0, stream>>>(phq, phk, vb, pst, pz, attn);

  // output projection
  k_f32_to_bf16<<<NTOK * Ec / T, T, 0, stream>>>(attn, Xbf, NTOK * Ec);
  k_gemm_xwt<<<gemm_blocks, T, 0, stream>>>(Xbf, W0b, b0, out, NTOK, Ec, Ec);
}
